// LSTMN_58944131170540
// MI455X (gfx1250) — matrix operand order repaired and checked by reading, not compiled, not run on hardware
//
#include <hip/hip_runtime.h>
#include <cstdint>

#define SEQ    256
#define BATCH  128
#define INDIM  512
#define HID    512
#define LAYERS 2
#define TAPE   30
#define GATES  (4*HID)   // 2048
#define BH     (BATCH*HID)

typedef __attribute__((ext_vector_type(16))) __bf16        v16bf;
typedef __attribute__((ext_vector_type(8)))  float         v8f;
typedef __attribute__((ext_vector_type(4)))  float         v4f;
typedef __attribute__((ext_vector_type(8)))  unsigned short v8u16;
typedef __attribute__((ext_vector_type(16))) unsigned short v16u16;

__device__ __forceinline__ unsigned short f2bf(float f) {
  unsigned int u = __float_as_uint(f);
  u += 0x7fffu + ((u >> 16) & 1u);          // round-to-nearest-even
  return (unsigned short)(u >> 16);
}

__device__ __forceinline__ float sigf(float x) { return 1.0f / (1.0f + __expf(-x)); }

// A fragment (16x32 bf16) from an f32 row-major matrix, converting in-register.
// ISA 7.12.2 16-bit A 16x32: lanes 0-15 hold M with K k0..k0+7 in halves 0..7 and
// k0+16..k0+23 in halves 8..15; k0 = kBase (lanes<16) or kBase+8 (lanes>=16).
__device__ __forceinline__ v16bf load_a_f32(const float* __restrict__ A, int lda,
                                            int mBase, int kBase, int lane) {
  const float* p = A + (size_t)(mBase + (lane & 15)) * lda + kBase + ((lane >> 4) << 3);
  v4f a0 = *(const v4f*)(p);
  v4f a1 = *(const v4f*)(p + 4);
  v4f a2 = *(const v4f*)(p + 16);
  v4f a3 = *(const v4f*)(p + 20);
  v16u16 r;
#pragma unroll
  for (int i = 0; i < 4; ++i) {
    r[i]      = f2bf(a0[i]);
    r[4 + i]  = f2bf(a1[i]);
    r[8 + i]  = f2bf(a2[i]);
    r[12 + i] = f2bf(a3[i]);
  }
  return __builtin_bit_cast(v16bf, r);
}

// B fragment (32x16 bf16): B(k,n) = W[nBase+n, k], W row-major (out, K).
// Lane n = lane&15, K = k0..k0+7 then k0+16..k0+23, k0 = kBase + (lane<16 ? 0 : 8).
__device__ __forceinline__ v16bf load_b_bf16(const unsigned short* __restrict__ W, int ldw,
                                             int nBase, int kBase, int lane) {
  const unsigned short* p = W + (size_t)(nBase + (lane & 15)) * ldw + kBase + ((lane >> 4) << 3);
  v8u16 b0 = *(const v8u16*)(p);
  v8u16 b1 = *(const v8u16*)(p + 16);
  v16u16 r;
#pragma unroll
  for (int i = 0; i < 8; ++i) { r[i] = b0[i]; r[8 + i] = b1[i]; }
  return __builtin_bit_cast(v16bf, r);
}

#define WMMA_BF16(A, B, C) \
  __builtin_amdgcn_wmma_f32_16x16x32_bf16(false, (A), false, (B), (short)0, (C), false, false)

// ---------------------------------------------------------------------------
// Fused LSTM gate GEMM + activations, split-K x4 through LDS.
// Block = 512 threads = 16 waves = 4 tiles x 4 K-slices.
// Slice 0/1: X @ W_ih^T halves; slice 2/3: H @ W_hh^T halves.
// Software-pipelined: next iteration's 5 fragments issued before the WMMAs.
// grid.x = 64 (256 tiles), grid.y = T.
// ---------------------------------------------------------------------------
__global__ void __launch_bounds__(512) gates_kernel(
    const float* __restrict__ A1, int a1_tstride,
    const float* __restrict__ A2,
    const unsigned short* __restrict__ Wih,
    const unsigned short* __restrict__ Whh,
    const float* __restrict__ bih, const float* __restrict__ bhh,
    const float* __restrict__ c_in,
    float* __restrict__ h_out, int ho_tstride,
    float* __restrict__ c_out) {
  __shared__ float red[3 * 4 * 4 * 256];            // 48 KB: slice-1..3 x tile x gate x 256

  int lane  = threadIdx.x & 31;
  int w     = threadIdx.x >> 5;                     // 0..15
  int tib   = w & 3;                                // tile in block
  int slice = w >> 2;                               // 0..3
  int tile  = blockIdx.x * 4 + tib;                 // 0..255
  int mBase = (tile >> 5) << 4;                     // batch tile
  int hBase = (tile & 31) << 4;                     // hidden tile

  const float* a1 = A1 + (size_t)blockIdx.y * a1_tstride;
  float* ho = h_out + (size_t)blockIdx.y * ho_tstride;
  float* co = c_out + (size_t)blockIdx.y * ho_tstride;

  v8f acc[4] = {};

  bool active = (slice < 2) || (A2 != nullptr);
  if (active) {
    const float* Amat          = (slice < 2) ? a1  : A2;
    const unsigned short* Wmat = (slice < 2) ? Wih : Whh;
    int kb = (slice & 1) * 256;

    v16bf a  = load_a_f32(Amat, 512, mBase, kb, lane);
    v16bf b0 = load_b_bf16(Wmat, 512, hBase + 0 * HID, kb, lane);
    v16bf b1 = load_b_bf16(Wmat, 512, hBase + 1 * HID, kb, lane);
    v16bf b2 = load_b_bf16(Wmat, 512, hBase + 2 * HID, kb, lane);
    v16bf b3 = load_b_bf16(Wmat, 512, hBase + 3 * HID, kb, lane);
#pragma unroll
    for (int i = 0; i < 8; ++i) {
      v16bf an = a, bn0 = b0, bn1 = b1, bn2 = b2, bn3 = b3;
      if (i < 7) {                                  // prefetch next K-step
        an  = load_a_f32(Amat, 512, mBase, kb + 32, lane);
        bn0 = load_b_bf16(Wmat, 512, hBase + 0 * HID, kb + 32, lane);
        bn1 = load_b_bf16(Wmat, 512, hBase + 1 * HID, kb + 32, lane);
        bn2 = load_b_bf16(Wmat, 512, hBase + 2 * HID, kb + 32, lane);
        bn3 = load_b_bf16(Wmat, 512, hBase + 3 * HID, kb + 32, lane);
      }
      acc[0] = WMMA_BF16(a, b0, acc[0]);
      acc[1] = WMMA_BF16(a, b1, acc[1]);
      acc[2] = WMMA_BF16(a, b2, acc[2]);
      acc[3] = WMMA_BF16(a, b3, acc[3]);
      a = an; b0 = bn0; b1 = bn1; b2 = bn2; b3 = bn3;
      kb += 32;
    }
  }

  if (slice > 0) {                                  // park partials in LDS
#pragma unroll
    for (int g = 0; g < 4; ++g) {
      float* dst = &red[((((slice - 1) * 4 + tib) * 4 + g) << 8) + (lane << 3)];
      *(v8f*)dst = acc[g];
    }
  }
  __syncthreads();
  if (slice != 0) return;

#pragma unroll
  for (int s = 1; s < 4; ++s) {
#pragma unroll
    for (int g = 0; g < 4; ++g) {
      const float* src = &red[((((s - 1) * 4 + tib) * 4 + g) << 8) + (lane << 3)];
      acc[g] += *(const v8f*)src;
    }
  }

  // Epilogue. C/D layout: VGPR r -> row mBase + r + (lane>=16 ? 8:0), col = hBase + (lane&15).
  int col    = hBase + (lane & 15);
  int rowOff = mBase + ((lane >> 4) << 3);
  float bI = bih[0 * HID + col] + bhh[0 * HID + col];
  float bF = bih[1 * HID + col] + bhh[1 * HID + col];
  float bG = bih[2 * HID + col] + bhh[2 * HID + col];
  float bO = bih[3 * HID + col] + bhh[3 * HID + col];
#pragma unroll
  for (int r = 0; r < 8; ++r) {
    int row = rowOff + r;
    float iv = sigf(acc[0][r] + bI);
    float fv = sigf(acc[1][r] + bF);
    float gv = tanhf(acc[2][r] + bG);
    float ov = sigf(acc[3][r] + bO);
    float cprev = (c_in != nullptr) ? c_in[(size_t)row * HID + col] : 0.0f;
    float cn = fv * cprev + iv * gv;
    float hn = ov * tanhf(cn);
    co[(size_t)row * HID + col] = cn;
    ho[(size_t)row * HID + col] = hn;
  }
}

// ---------------------------------------------------------------------------
// x_attn[l,b,h] = sum_i attn_wx[l,h,i] * x[b,i]
// Block = 512 threads = 16 waves = 8 tiles x 2 K-slices, LDS reduce. grid 64.
// ---------------------------------------------------------------------------
__global__ void __launch_bounds__(512) xattn_kernel(
    const float* __restrict__ x, const unsigned short* __restrict__ wxb,
    float* __restrict__ xa) {
  __shared__ float red[8 * 256];                    // 8 KB

  int lane  = threadIdx.x & 31;
  int w     = threadIdx.x >> 5;                     // 0..15
  int tib   = w >> 1;                               // 0..7
  int slice = w & 1;
  int tile  = blockIdx.x * 8 + tib;                 // 0..511
  int l     = tile >> 8;
  int mBase = ((tile >> 5) & 7) << 4;
  int nBase = (tile & 31) << 4;
  const unsigned short* W = wxb + (size_t)l * HID * INDIM;

  int kb = slice * 256;
  v8f acc = {};
  v16bf a = load_a_f32(x, INDIM, mBase, kb, lane);
  v16bf b = load_b_bf16(W, INDIM, nBase, kb, lane);
#pragma unroll
  for (int i = 0; i < 8; ++i) {
    v16bf an = a, bn = b;
    if (i < 7) {
      an = load_a_f32(x, INDIM, mBase, kb + 32, lane);
      bn = load_b_bf16(W, INDIM, nBase, kb + 32, lane);
    }
    acc = WMMA_BF16(a, b, acc);
    a = an; b = bn; kb += 32;
  }

  if (slice == 1) *(v8f*)&red[(tib << 8) + (lane << 3)] = acc;
  __syncthreads();
  if (slice != 0) return;
  acc += *(const v8f*)&red[(tib << 8) + (lane << 3)];

  int col    = nBase + (lane & 15);
  int rowOff = mBase + ((lane >> 4) << 3);
  float* out = xa + (size_t)l * BH;
#pragma unroll
  for (int r = 0; r < 8; ++r) out[(size_t)(rowOff + r) * HID + col] = acc[r];
}

// ---------------------------------------------------------------------------
// score[t,l,b] = sum_h tanh(h_tape*wh_d + x_attn + prev_h*wht_d) * v
// One wave per (t,l,b): 7680 waves, wave-level shuffle reduction.
// ---------------------------------------------------------------------------
__global__ void __launch_bounds__(256) score_kernel(
    const float* __restrict__ h_tape, const float* __restrict__ x_attn,
    const float* __restrict__ prev_h, const float* __restrict__ whd,
    const float* __restrict__ whtd, const float* __restrict__ v,
    float* __restrict__ score) {
  int lane = threadIdx.x & 31;
  int wave = blockIdx.x * 8 + (threadIdx.x >> 5);   // 0..7679
  int t = wave >> 8;
  int l = (wave >> 7) & 1;
  int b = wave & 127;
  const float* ht  = h_tape + (size_t)((t * 2 + l) * BATCH + b) * HID;
  const float* xa  = x_attn + (size_t)(l * BATCH + b) * HID;
  const float* ph  = prev_h + (size_t)(l * BATCH + b) * HID;
  const float* wd  = whd + l * HID;
  const float* wtd = whtd + l * HID;
  const float* vv  = v + l * HID;
  float s = 0.0f;
#pragma unroll
  for (int j = 0; j < 16; ++j) {
    int h = lane + (j << 5);
    float a = tanhf(ht[h] * wd[h] + xa[h] + ph[h] * wtd[h]);
    s += a * vv[h];
  }
  for (int o = 16; o > 0; o >>= 1) s += __shfl_down(s, o, 32);
  if (lane == 0) score[(t * 2 + l) * BATCH + b] = s;
}

// ---------------------------------------------------------------------------
// Softmax over tape + weighted sums: prev_h/prev_c = sum_t alpha_t * tape_t.
// One block per (l,b); tape-slot order is irrelevant (softmax is a set op).
// ---------------------------------------------------------------------------
__global__ void __launch_bounds__(256) combine_kernel(
    const float* __restrict__ score, const float* __restrict__ h_tape,
    const float* __restrict__ c_tape, float* __restrict__ prev_h,
    float* __restrict__ prev_c) {
  int lb = blockIdx.x;
  int l = lb >> 7, b = lb & 127;
  float sc[TAPE];
  float mx = -3.0e38f;
#pragma unroll
  for (int t = 0; t < TAPE; ++t) {
    sc[t] = score[(t * 2 + l) * BATCH + b];
    mx = fmaxf(mx, sc[t]);
  }
  float ssum = 0.0f;
#pragma unroll
  for (int t = 0; t < TAPE; ++t) { sc[t] = __expf(sc[t] - mx); ssum += sc[t]; }
  float inv = 1.0f / ssum;
  for (int h = threadIdx.x; h < HID; h += 256) {
    float ah = 0.0f, ac = 0.0f;
#pragma unroll
    for (int t = 0; t < TAPE; ++t) {
      size_t idx = (size_t)((t * 2 + l) * BATCH + b) * HID + h;
      ah += sc[t] * h_tape[idx];
      ac += sc[t] * c_tape[idx];
    }
    size_t o = (size_t)(l * BATCH + b) * HID + h;
    prev_h[o] = ah * inv;
    prev_c[o] = ac * inv;
  }
}

// ---------------------------------------------------------------------------
// Per-call prep: bf16 weight copies, attn diagonals, zero prev_h.
// ---------------------------------------------------------------------------
__global__ void prep_kernel(const float* __restrict__ Wih, const float* __restrict__ Whh,
                            const float* __restrict__ wx, const float* __restrict__ wh,
                            const float* __restrict__ wht,
                            unsigned short* __restrict__ Wih_b,
                            unsigned short* __restrict__ Whh_b,
                            unsigned short* __restrict__ wx_b,
                            float* __restrict__ whd, float* __restrict__ whtd,
                            float* __restrict__ prev_h) {
  int idx = blockIdx.x * blockDim.x + threadIdx.x;
  if (idx < LAYERS * GATES * INDIM) {
    Wih_b[idx] = f2bf(Wih[idx]);
    Whh_b[idx] = f2bf(Whh[idx]);
  }
  if (idx < LAYERS * HID * INDIM) wx_b[idx] = f2bf(wx[idx]);
  if (idx < LAYERS * HID) {
    int l = idx >> 9, h = idx & 511;
    whd[idx]  = wh[((size_t)(l * HID + h)) * HID + h];
    whtd[idx] = wht[((size_t)(l * HID + h)) * HID + h];
  }
  if (idx < LAYERS * BH) prev_h[idx] = 0.0f;
}

__global__ void output_kernel(const float* __restrict__ h_tape, float* __restrict__ out,
                              int start) {
  int idx = blockIdx.x * blockDim.x + threadIdx.x;
  if (idx >= TAPE * BH) return;
  int i = idx / BH;
  int rem = idx - i * BH;
  int slot = (start + i) % TAPE;
  out[idx] = h_tape[(size_t)(slot * 2 + (LAYERS - 1)) * BH + rem];
}

extern "C" void kernel_launch(void* const* d_in, const int* in_sizes, int n_in,
                              void* d_out, int out_size, void* d_ws, size_t ws_size,
                              hipStream_t stream) {
  const float* xs       = (const float*)d_in[0];
  const float* W_ih     = (const float*)d_in[1];
  const float* W_hh     = (const float*)d_in[2];
  const float* b_ih     = (const float*)d_in[3];
  const float* b_hh     = (const float*)d_in[4];
  const float* attn_wh  = (const float*)d_in[5];
  const float* attn_wx  = (const float*)d_in[6];
  const float* attn_wht = (const float*)d_in[7];
  const float* attn_v   = (const float*)d_in[8];

  char* ws = (char*)d_ws;
  size_t off = 0;
  auto alloc = [&](size_t bytes) -> void* {
    void* p = ws + off;
    off = (off + bytes + 255) & ~(size_t)255;
    return p;
  };
  unsigned short* Wih_b = (unsigned short*)alloc((size_t)LAYERS * GATES * INDIM * 2);
  unsigned short* Whh_b = (unsigned short*)alloc((size_t)LAYERS * GATES * INDIM * 2);
  unsigned short* wx_b  = (unsigned short*)alloc((size_t)LAYERS * HID * INDIM * 2);
  float* h_tape = (float*)alloc((size_t)TAPE * LAYERS * BH * 4);
  float* c_tape = (float*)alloc((size_t)TAPE * LAYERS * BH * 4);
  float* x_attn = (float*)alloc((size_t)LAYERS * BH * 4);
  float* prev_h = (float*)alloc((size_t)LAYERS * BH * 4);
  float* prev_c = (float*)alloc((size_t)LAYERS * BH * 4);
  float* score  = (float*)alloc((size_t)TAPE * LAYERS * BATCH * 4);
  float* whd    = (float*)alloc((size_t)LAYERS * HID * 4);
  float* whtd   = (float*)alloc((size_t)LAYERS * HID * 4);

  prep_kernel<<<(LAYERS * GATES * INDIM + 255) / 256, 256, 0, stream>>>(
      W_ih, W_hh, attn_wx, attn_wh, attn_wht, Wih_b, Whh_b, wx_b, whd, whtd, prev_h);

  // Phase 1: fill the tape — 30 independent steps from zero state.
  dim3 g1(64, TAPE);
  gates_kernel<<<g1, 512, 0, stream>>>(xs, BATCH * INDIM, nullptr,
      Wih_b, Whh_b, b_ih, b_hh, nullptr,
      h_tape, LAYERS * BH, c_tape);
  gates_kernel<<<g1, 512, 0, stream>>>(h_tape, LAYERS * BH, nullptr,
      Wih_b + (size_t)GATES * INDIM, Whh_b + (size_t)GATES * INDIM,
      b_ih + GATES, b_hh + GATES, nullptr,
      h_tape + BH, LAYERS * BH, c_tape + BH);

  // Phase 2: 226 sequential scan steps.
  for (int k = 0; k < SEQ - TAPE; ++k) {
    const float* x_t = xs + (size_t)(TAPE + k) * BATCH * INDIM;
    int slot = k % TAPE;
    float* h_slot = h_tape + (size_t)(slot * 2) * BH;
    float* c_slot = c_tape + (size_t)(slot * 2) * BH;

    xattn_kernel<<<64, 512, 0, stream>>>(x_t, wx_b, x_attn);
    score_kernel<<<960, 256, 0, stream>>>(h_tape, x_attn, prev_h, whd, whtd, attn_v, score);
    combine_kernel<<<256, 256, 0, stream>>>(score, h_tape, c_tape, prev_h, prev_c);

    // layer 0 (overwrites oldest tape slot)
    gates_kernel<<<dim3(64, 1), 512, 0, stream>>>(x_t, 0, prev_h,
        Wih_b, Whh_b, b_ih, b_hh, prev_c,
        h_slot, 0, c_slot);
    // layer 1 (input = layer-0 h just written into the slot)
    gates_kernel<<<dim3(64, 1), 512, 0, stream>>>(h_slot, 0, prev_h + BH,
        Wih_b + (size_t)GATES * INDIM, Whh_b + (size_t)GATES * INDIM,
        b_ih + GATES, b_hh + GATES, prev_c + BH,
        h_slot + BH, 0, c_slot + BH);
  }

  int n = TAPE * BH;
  output_kernel<<<(n + 255) / 256, 256, 0, stream>>>(h_tape, (float*)d_out,
                                                     (SEQ - TAPE) % TAPE);
}
